// DenseNetWithPCAM_42588895707592
// MI455X (gfx1250) — matrix operand order repaired and checked by reading, not compiled, not run on hardware
//
#include <hip/hip_runtime.h>
#include <hip/hip_bf16.h>

#define B_  256
#define C_  1024
#define HW_ 49
#define K_  50176   // C_*HW_
#define DH  768
#define NC  5

typedef __attribute__((ext_vector_type(16))) __bf16       v16bf;
typedef __attribute__((ext_vector_type(8)))  float        v8f;
typedef __attribute__((ext_vector_type(4)))  unsigned int v4u;
typedef __attribute__((ext_vector_type(8)))  int          v8i;
typedef __attribute__((ext_vector_type(4)))  int          v4i;

#if defined(__HIP_DEVICE_COMPILE__) && __has_builtin(__builtin_amdgcn_tensor_load_to_lds)
#define HAS_TDM 1
#else
#define HAS_TDM 0
#endif

// ---------------------------------------------------------------------------
// TDM: DMA a 2D f32 tile (32 cols x `tileRows` rows, row stride = K_ elems)
// from global memory into LDS, padding each 128B row to 144B (36 f32).
// Must be executed by a single wave (EXEC is ignored by TENSOR ops).
// ---------------------------------------------------------------------------
__device__ __forceinline__ unsigned lds_off(const void* p) {
  // generic pointer to LDS: low 32 bits of the flat address are the LDS offset
  return (unsigned)(unsigned long long)(uintptr_t)p;
}

#if HAS_TDM
__device__ __forceinline__ void tdm_load_tile(unsigned lds_byte, const float* gptr,
                                              unsigned tileRows) {
  const unsigned long long ga = (unsigned long long)(uintptr_t)gptr;
  v4u g0;
  g0[0] = 1u;                                        // count=1, user descriptor
  g0[1] = lds_byte;                                  // lds_addr
  g0[2] = (unsigned)(ga & 0xffffffffu);              // global_addr lo
  g0[3] = (unsigned)((ga >> 32) & 0x01ffffffu)       // global_addr hi (57-bit)
        | (2u << 30);                                // type = 2 ("image")
  const unsigned td0 = (unsigned)K_;                 // tensor_dim0 (elements)
  v8i g1;
  g1[0] = (int)((2u << 16)      // data_size = 4B
        | (1u << 20)            // pad_enable
        | (4u << 22)            // pad_interval: 32 DWORDs (one 128B row)
        | (3u << 25));          // pad_amount:   4 DWORDs (16B) -> 144B pitch
  g1[1] = (int)((td0 & 0xffffu) << 16);              // tensor_dim0[15:0]
  g1[2] = (int)((td0 >> 16) | ((tileRows & 0xffffu) << 16)); // dim0 hi | tensor_dim1 lo
  g1[3] = (int)((tileRows >> 16) | (32u << 16));     // tensor_dim1 hi | tile_dim0 = 32
  g1[4] = (int)(tileRows & 0xffffu);                 // tile_dim1 ; tile_dim2 = 0
  g1[5] = (int)K_;                                   // tensor_dim0_stride lo32
  g1[6] = 0;                                         // stride hi | dim1_stride lo
  g1[7] = 0;
  const v4i gz = {0, 0, 0, 0};
#if __clang_major__ >= 23
  const v8i gz8 = {0, 0, 0, 0, 0, 0, 0, 0};
  __builtin_amdgcn_tensor_load_to_lds(g0, g1, gz, gz, gz8, 0);
#else
  __builtin_amdgcn_tensor_load_to_lds(g0, g1, gz, gz, 0);
#endif
}
#endif

// ---------------------------------------------------------------------------
// Kernel 1: per-(b,c) spatial max -> ch_max, and stream features copy to out
// ---------------------------------------------------------------------------
__global__ __launch_bounds__(256)
void pcam_kernel(const float* __restrict__ feat, float* __restrict__ chmax,
                 float* __restrict__ feat_out) {
  const int pair = blockIdx.x * 8 + (threadIdx.x >> 5);
  const int lane = threadIdx.x & 31;
  const size_t base = (size_t)pair * HW_;

  float v0 = feat[base + lane];
  float m  = v0;
  float v1 = 0.f;
  if (lane < 17) { v1 = feat[base + 32 + lane]; m = fmaxf(m, v1); }
  #pragma unroll
  for (int off = 16; off > 0; off >>= 1)
    m = fmaxf(m, __shfl_xor(m, off, 32));

  if (lane == 0) chmax[pair] = m;
  feat_out[base + lane] = v0;
  if (lane < 17) feat_out[base + 32 + lane] = v1;
}

// ---------------------------------------------------------------------------
// Kernel 2: Cacc[m][n] = b1[n]
// ---------------------------------------------------------------------------
__global__ __launch_bounds__(256)
void init_acc_kernel(float* __restrict__ Cacc, const float* __restrict__ b1) {
  int i = blockIdx.x * 256 + threadIdx.x;
  if (i < B_ * DH) Cacc[i] = b1[i % DH];
}

// ---------------------------------------------------------------------------
// Kernel 3: split-K WMMA GEMM: Cacc += (feat * chmax) @ W1^T
//   grid.x = 6 N-blocks of 128, grid.y = 32 K-chunks of 1568 (49 steps of 32,
//   each chunk covers exactly 32 channels). TDM double-buffers both tiles.
// ---------------------------------------------------------------------------
#define APF 36   // f32 LDS pitch for A rows (128B row + 16B TDM pad)
#define BPF 36   // f32 LDS pitch for B rows
#define CHP 36   // f32 LDS pitch for chmax slab

__global__ __launch_bounds__(256)
void gemm1_kernel(const float* __restrict__ feat, const float* __restrict__ W1,
                  const float* __restrict__ chmax, float* __restrict__ Cacc) {
  __shared__ float smA[2][256 * APF];   // 2 x 36 KB
  __shared__ float smB[2][128 * BPF];   // 2 x 18 KB
  __shared__ float smCh[256 * CHP];     // 36 KB

  const int tid  = threadIdx.x;
  const int lane = tid & 31;
  const int wave = tid >> 5;
  const int nb   = blockIdx.x;          // 0..5
  const int kc   = blockIdx.y;          // 0..31

  // ---- stage ch_max slab: 256 rows x 32 channels for this K-chunk ----
  {
    const float4* src = (const float4*)(chmax + (size_t)tid * C_ + kc * 32);
    float4* dst = (float4*)&smCh[tid * CHP];
    #pragma unroll
    for (int q = 0; q < 8; ++q) dst[q] = src[q];
  }

  const v8f vzero = {0.f,0.f,0.f,0.f,0.f,0.f,0.f,0.f};
  v8f acc[16];
  #pragma unroll
  for (int i = 0; i < 16; ++i) acc[i] = vzero;

  const float* Abase = feat + (size_t)kc * 1568;                      // row 0
  const float* Bbase = W1 + (size_t)(nb * 128) * K_ + (size_t)kc * 1568;

#if HAS_TDM
  if (wave == 0) {
    tdm_load_tile(lds_off(&smA[0][0]), Abase, 256);
    tdm_load_tile(lds_off(&smB[0][0]), Bbase, 128);
  }
  __builtin_amdgcn_s_wait_tensorcnt(0);
#else
  {
    const float4* s = (const float4*)(Abase + (size_t)tid * K_);
    float4* d = (float4*)&smA[0][tid * APF];
    #pragma unroll
    for (int q = 0; q < 8; ++q) d[q] = s[q];
    const float4* s2 = (const float4*)(Bbase + (size_t)(tid >> 1) * K_ + (tid & 1) * 16);
    float4* d2 = (float4*)&smB[0][(tid >> 1) * BPF + (tid & 1) * 16];
    #pragma unroll
    for (int q = 0; q < 4; ++q) d2[q] = s2[q];
  }
#endif
  __syncthreads();

  for (int step = 0; step < 49; ++step) {
    const int cur = step & 1;
    const int nxt = cur ^ 1;

    // ---- prefetch next tiles via TDM while we compute on current ----
    if (step < 48) {
#if HAS_TDM
      if (wave == 0) {
        tdm_load_tile(lds_off(&smA[nxt][0]), Abase + (step + 1) * 32, 256);
        tdm_load_tile(lds_off(&smB[nxt][0]), Bbase + (step + 1) * 32, 128);
      }
#else
      const float4* s = (const float4*)(Abase + (step + 1) * 32 + (size_t)tid * K_);
      float4* d = (float4*)&smA[nxt][tid * APF];
      #pragma unroll
      for (int q = 0; q < 8; ++q) d[q] = s[q];
      const float4* s2 = (const float4*)(Bbase + (step + 1) * 32 +
                                         (size_t)(tid >> 1) * K_ + (tid & 1) * 16);
      float4* d2 = (float4*)&smB[nxt][(tid >> 1) * BPF + (tid & 1) * 16];
      #pragma unroll
      for (int q = 0; q < 4; ++q) d2[q] = s2[q];
#endif
    }

    // ---- channel-scale scalars for this 32-k slab (SALU only) ----
    const int k0c  = step * 32;                 // k offset within chunk
    const int cc0  = k0c / HW_;                 // channel within chunk (0..31)
    const int bdry = (cc0 + 1) * HW_ - k0c;     // 1..49
    const int cc1  = (bdry < 32) ? cc0 + 1 : cc0;

    const float* bufA = smA[cur];
    const float* bufB = smB[cur];

    // ---- A fragments: f32 LDS -> scale by ch_max -> bf16 ----
    v16bf afr[2], bfr[8];
    const int ar  = lane & 15;
    const int ak0 = (lane >> 4) << 3;           // lanes 0-15: K{0,16}; 16-31: K{8,24}
    #pragma unroll
    for (int mi = 0; mi < 2; ++mi) {
      const int row = (wave * 2 + mi) * 16 + ar;
      const float* rp = &bufA[row * APF + ak0];
      const float mx0 = smCh[row * CHP + cc0];
      const float mx1 = smCh[row * CHP + cc1];
      float4 x0 = ((const float4*)rp)[0];
      float4 x1 = ((const float4*)rp)[1];
      float4 x2 = ((const float4*)(rp + 16))[0];
      float4 x3 = ((const float4*)(rp + 16))[1];
#define CVA(idx, val, kk) afr[mi][idx] = (__bf16)((val) * (((kk) < bdry) ? mx0 : mx1))
      CVA(0,  x0.x, ak0+0);  CVA(1,  x0.y, ak0+1);
      CVA(2,  x0.z, ak0+2);  CVA(3,  x0.w, ak0+3);
      CVA(4,  x1.x, ak0+4);  CVA(5,  x1.y, ak0+5);
      CVA(6,  x1.z, ak0+6);  CVA(7,  x1.w, ak0+7);
      CVA(8,  x2.x, ak0+16); CVA(9,  x2.y, ak0+17);
      CVA(10, x2.z, ak0+18); CVA(11, x2.w, ak0+19);
      CVA(12, x3.x, ak0+20); CVA(13, x3.y, ak0+21);
      CVA(14, x3.z, ak0+22); CVA(15, x3.w, ak0+23);
#undef CVA
    }

    // ---- B fragments: f32 LDS -> bf16 (same K order as A) ----
    const int bk0 = (lane >> 4) << 3;           // lanes 0-15: K{0,16}; 16-31: K{8,24}
    #pragma unroll
    for (int ni = 0; ni < 8; ++ni) {
      const int row = ni * 16 + ar;
      const float4* q = (const float4*)&bufB[row * BPF + bk0];
      float4 y0 = q[0], y1 = q[1], y2 = q[4], y3 = q[5];
      bfr[ni][0]  = (__bf16)y0.x; bfr[ni][1]  = (__bf16)y0.y;
      bfr[ni][2]  = (__bf16)y0.z; bfr[ni][3]  = (__bf16)y0.w;
      bfr[ni][4]  = (__bf16)y1.x; bfr[ni][5]  = (__bf16)y1.y;
      bfr[ni][6]  = (__bf16)y1.z; bfr[ni][7]  = (__bf16)y1.w;
      bfr[ni][8]  = (__bf16)y2.x; bfr[ni][9]  = (__bf16)y2.y;
      bfr[ni][10] = (__bf16)y2.z; bfr[ni][11] = (__bf16)y2.w;
      bfr[ni][12] = (__bf16)y3.x; bfr[ni][13] = (__bf16)y3.y;
      bfr[ni][14] = (__bf16)y3.z; bfr[ni][15] = (__bf16)y3.w;
    }

    #pragma unroll
    for (int mi = 0; mi < 2; ++mi)
      #pragma unroll
      for (int ni = 0; ni < 8; ++ni)
        acc[mi * 8 + ni] = __builtin_amdgcn_wmma_f32_16x16x32_bf16(
            false, afr[mi], false, bfr[ni],
            (short)0, acc[mi * 8 + ni], false, false);

#if HAS_TDM
    __builtin_amdgcn_s_wait_tensorcnt(0);   // next-step DMA landed (wave 0)
#endif
    __syncthreads();
  }

  // ---- split-K merge: atomic f32 add into Cacc [256 x 768] ----
  const int nrow = lane & 15;
  const int mhi  = (lane >> 4) * 8;
  #pragma unroll
  for (int mi = 0; mi < 2; ++mi) {
    #pragma unroll
    for (int ni = 0; ni < 8; ++ni) {
      const int ng = nb * 128 + ni * 16 + nrow;
      #pragma unroll
      for (int e = 0; e < 8; ++e) {
        const int mg = (wave * 2 + mi) * 16 + mhi + e;
        atomicAdd(&Cacc[mg * DH + ng], acc[mi * 8 + ni][e]);
      }
    }
  }
}

// ---------------------------------------------------------------------------
// Kernel 4: logits[m][n] = relu(Cacc[m]) . W2[n] + b2[n]
// ---------------------------------------------------------------------------
__global__ __launch_bounds__(256)
void gemm2_kernel(const float* __restrict__ Cacc, const float* __restrict__ W2,
                  const float* __restrict__ b2, float* __restrict__ logits) {
  __shared__ float red[NC][256];
  const int m = blockIdx.x;
  const int t = threadIdx.x;

  float p[NC] = {0.f, 0.f, 0.f, 0.f, 0.f};
  for (int k = t; k < DH; k += 256) {
    const float h = fmaxf(Cacc[m * DH + k], 0.f);
    #pragma unroll
    for (int n = 0; n < NC; ++n) p[n] += h * W2[n * DH + k];
  }
  #pragma unroll
  for (int n = 0; n < NC; ++n) red[n][t] = p[n];
  __syncthreads();
  for (int s = 128; s > 0; s >>= 1) {
    if (t < s) {
      #pragma unroll
      for (int n = 0; n < NC; ++n) red[n][t] += red[n][t + s];
    }
    __syncthreads();
  }
  if (t < NC) logits[m * NC + t] = red[t][0] + b2[t];
}

// ---------------------------------------------------------------------------
extern "C" void kernel_launch(void* const* d_in, const int* in_sizes, int n_in,
                              void* d_out, int out_size, void* d_ws, size_t ws_size,
                              hipStream_t stream) {
  const float* feat = (const float*)d_in[0];
  const float* W1   = (const float*)d_in[1];
  const float* b1   = (const float*)d_in[2];
  const float* W2   = (const float*)d_in[3];
  const float* b2   = (const float*)d_in[4];

  float* out   = (float*)d_out;
  float* chmax = (float*)d_ws;
  float* Cacc  = chmax + (size_t)B_ * C_;

  pcam_kernel<<<(B_ * C_) / 8, 256, 0, stream>>>(feat, chmax, out + B_ * NC);
  init_acc_kernel<<<(B_ * DH + 255) / 256, 256, 0, stream>>>(Cacc, b1);
  gemm1_kernel<<<dim3(6, 32, 1), 256, 0, stream>>>(feat, W1, chmax, Cacc);
  gemm2_kernel<<<B_, 256, 0, stream>>>(Cacc, W2, b2, out);
}
